// Seq2Seq_19937238188046
// MI455X (gfx1250) — matrix operand order repaired and checked by reading, not compiled, not run on hardware
//
#include <hip/hip_runtime.h>
#include <hip/hip_bf16.h>

// ---------------- problem constants ----------------
#define Bc  32
#define Sc  64
#define Tc  48
#define Ec  128
#define Hc  256
#define HDc 512
#define Vc  32001

typedef unsigned short ushort_t;
typedef __attribute__((ext_vector_type(16))) __bf16 v16bf;
typedef __attribute__((ext_vector_type(8)))  float  v8f;

union FragU { v16bf v; uint4 q[2]; };

__device__ __forceinline__ ushort_t f2bf(float f) {
    unsigned int u = __float_as_uint(f);
    unsigned int r = (u + 0x7FFFu + ((u >> 16) & 1u)) >> 16;   // RNE
    return (ushort_t)r;
}
__device__ __forceinline__ float bf2f(ushort_t b) {
    return __uint_as_float(((unsigned int)b) << 16);
}
__device__ __forceinline__ float sigf(float x) {
    return 1.0f / (1.0f + expf(-x));
}

// ---------------- generic bf16 WMMA GEMM:  C(M,N) = A(M,K) * W(N,K)^T (+bias) ----------------
// One wave computes a 32x64 macro-tile: 2 A fragments x 4 B fragments -> 8 WMMAs per K-step.
// K is a compile-time template parameter: the K-loop fully unrolls and all load offsets
// become instruction immediates (no per-iteration address math, no loop SALU).
// Out-of-range N columns are handled by CLAMPING the W row (loads always valid and
// convergent -> no EXEC manipulation in the hot loop); the store skips n >= N.
template <int K>
__global__ __launch_bounds__(256) void gemm_bf16_kernel(
    const ushort_t* __restrict__ A, int lda,
    const ushort_t* __restrict__ W, int ldw,
    float* __restrict__ C, int ldc,
    int M, int N,
    const float* __restrict__ bias,
    int nTilesN, int remap)
{
    int wid  = (blockIdx.x * blockDim.x + threadIdx.x) >> 5;   // global wave id (wave32)
    int lane = threadIdx.x & 31;
    int totalTiles = (M >> 5) * nTilesN;
    if (wid >= totalTiles) return;                             // wave-uniform: EXEC stays all-1s

    int tM = (wid / nTilesN) << 5;
    int tN = (wid % nTilesN) << 6;
    int lo = lane & 15, hi = lane >> 4;

    v8f acc[2][4];
    v8f zz = 0.0f;
    #pragma unroll
    for (int i = 0; i < 2; ++i)
        #pragma unroll
        for (int j = 0; j < 4; ++j) acc[i][j] = zz;

    // A-matrix 16x32 layout: lane = M (mod 16); lanes 0-15 hold K {0..7,16..23}, lanes 16-31 {8..15,24..31}
    const ushort_t* Arow0 = A + (size_t)(tM + lo)      * (size_t)lda + (hi << 3);
    const ushort_t* Arow1 = A + (size_t)(tM + 16 + lo) * (size_t)lda + (hi << 3);
    // B-matrix 32x16 layout: lane = N (mod 16); lanes 0-15 hold K {0..7,16..23}, lanes 16-31 {8..15,24..31}
    const ushort_t* Wrow[4];
    #pragma unroll
    for (int j = 0; j < 4; ++j) {
        int n  = tN + (j << 4) + lo;
        int nc = (n < N) ? n : (N - 1);        // clamp: garbage column never stored
        Wrow[j] = W + (size_t)nc * (size_t)ldw + (hi << 3);
    }

    #pragma unroll
    for (int kb = 0; kb < K; kb += 32) {
        FragU a0, a1;
        a0.q[0] = *reinterpret_cast<const uint4*>(Arow0 + kb);
        a0.q[1] = *reinterpret_cast<const uint4*>(Arow0 + kb + 16);
        a1.q[0] = *reinterpret_cast<const uint4*>(Arow1 + kb);
        a1.q[1] = *reinterpret_cast<const uint4*>(Arow1 + kb + 16);
        #pragma unroll
        for (int j = 0; j < 4; ++j) {
            FragU b;
            b.q[0] = *reinterpret_cast<const uint4*>(Wrow[j] + kb);
            b.q[1] = *reinterpret_cast<const uint4*>(Wrow[j] + kb + 16);
            acc[0][j] = __builtin_amdgcn_wmma_f32_16x16x32_bf16(
                false, a0.v, false, b.v, (short)0, acc[0][j], false, false);
            acc[1][j] = __builtin_amdgcn_wmma_f32_16x16x32_bf16(
                false, a1.v, false, b.v, (short)0, acc[1][j], false, false);
        }
    }

    // C/D layout: VGPR r holds M = r + 8*(lane>=16), N = lane&15
    #pragma unroll
    for (int i = 0; i < 2; ++i) {
        #pragma unroll
        for (int j = 0; j < 4; ++j) {
            int n = tN + (j << 4) + lo;
            if (n >= N) continue;
            float bv = bias ? bias[n] : 0.0f;
            #pragma unroll
            for (int r = 0; r < 8; ++r) {
                int m = tM + (i << 4) + r + (hi << 3);
                float val = acc[i][j][r] + bv;
                // remap: A rows are (t*B + b); output rows must be (b*T + t)
                size_t orow = remap ? ((size_t)(m & (Bc - 1)) * Tc + (size_t)(m >> 5))
                                    : (size_t)m;
                C[orow * (size_t)ldc + (size_t)n] = val;
            }
        }
    }
}

// ---------------- elementwise / gather kernels ----------------
__global__ void f32_to_bf16_kernel(const float* __restrict__ src, ushort_t* __restrict__ dst, int n) {
    int i = blockIdx.x * blockDim.x + threadIdx.x;
    if (i < n) dst[i] = f2bf(src[i]);
}

// x_src (S,B,E) bf16 from emb_src[input_sequence]
__global__ void gather_src_kernel(const float* __restrict__ emb, const int* __restrict__ seq,
                                  ushort_t* __restrict__ dst) {
    int i = blockIdx.x * blockDim.x + threadIdx.x;
    if (i >= Sc * Bc * Ec) return;
    int e = i % Ec;
    int b = (i / Ec) % Bc;
    int s = i / (Ec * Bc);
    int tok = seq[b * Sc + s];
    dst[i] = f2bf(emb[(size_t)tok * Ec + e]);
}

// dec_in (T,B,E) bf16: t==0 -> 0, else emb_tgt[output_sequence[b, t-1]]
__global__ void gather_dec_kernel(const float* __restrict__ emb, const int* __restrict__ seq,
                                  ushort_t* __restrict__ dst) {
    int i = blockIdx.x * blockDim.x + threadIdx.x;
    if (i >= Tc * Bc * Ec) return;
    int e = i % Ec;
    int b = (i / Ec) % Bc;
    int t = i / (Ec * Bc);
    float val = 0.0f;
    if (t > 0) {
        int tok = seq[b * Tc + (t - 1)];
        val = emb[(size_t)tok * Ec + e];
    }
    dst[i] = f2bf(val);
}

// encoder LSTM cell (dir 0 = forward, 1 = backward with masked carry)
__global__ void lstm_cell_enc_kernel(const float* __restrict__ xg, const float* __restrict__ hg,
                                     float* __restrict__ h, float* __restrict__ c,
                                     ushort_t* __restrict__ h_bf, ushort_t* __restrict__ enc_bf,
                                     const int* __restrict__ lens, int s, int dir) {
    int idx = blockIdx.x * blockDim.x + threadIdx.x;
    if (idx >= Bc * Hc) return;
    int b = idx / Hc, u = idx % Hc;
    const float* xr = xg + (size_t)(s * Bc + b) * (4 * Hc);
    const float* hr = hg + (size_t)b * (4 * Hc);
    float gi = xr[u]          + hr[u];
    float gf = xr[Hc + u]     + hr[Hc + u];
    float gg = xr[2 * Hc + u] + hr[2 * Hc + u];
    float go = xr[3 * Hc + u] + hr[3 * Hc + u];
    float cn = sigf(gf) * c[idx] + sigf(gi) * tanhf(gg);
    float hn = sigf(go) * tanhf(cn);
    bool valid = s < lens[b];
    float hout, cout;
    if (dir == 0) { hout = hn; cout = cn; }
    else          { hout = valid ? hn : h[idx]; cout = valid ? cn : c[idx]; }
    h[idx] = hout; c[idx] = cout;
    h_bf[idx] = f2bf(hout);
    float eo = valid ? hn : 0.0f;
    enc_bf[(size_t)(s * Bc + b) * (2 * Hc) + (size_t)dir * Hc + u] = f2bf(eo);
}

// decoder LSTM cell
__global__ void lstm_cell_dec_kernel(const float* __restrict__ xg, const float* __restrict__ hg,
                                     float* __restrict__ h, float* __restrict__ c,
                                     ushort_t* __restrict__ h_bf, int t) {
    int idx = blockIdx.x * blockDim.x + threadIdx.x;
    if (idx >= Bc * HDc) return;
    int b = idx / HDc, u = idx % HDc;
    const float* xr = xg + (size_t)(t * Bc + b) * (4 * HDc);
    const float* hr = hg + (size_t)b * (4 * HDc);
    float gi = xr[u]           + hr[u];
    float gf = xr[HDc + u]     + hr[HDc + u];
    float gg = xr[2 * HDc + u] + hr[2 * HDc + u];
    float go = xr[3 * HDc + u] + hr[3 * HDc + u];
    float cn = sigf(gf) * c[idx] + sigf(gi) * tanhf(gg);
    float hn = sigf(go) * tanhf(cn);
    h[idx] = hn; c[idx] = cn;
    h_bf[idx] = f2bf(hn);
}

// attention + context for decoder step t: writes a_bf[(t,B,HD)] = bf16(h + ctx)
__global__ void attention_kernel(const float* __restrict__ kp, const float* __restrict__ qp,
                                 const ushort_t* __restrict__ enc_bf, const float* __restrict__ vv,
                                 const float* __restrict__ h_d, const int* __restrict__ lens,
                                 ushort_t* __restrict__ a_bf, int t) {
    int b = blockIdx.x;
    int tid = threadIdx.x;
    __shared__ float part[256];
    __shared__ float sh_e[Sc];
    __shared__ float sw[Sc];

    // energy[s] = v . tanh(kp[s,b,:] + qp[b,:]) ; 4 threads per s, 128 dims each
    int s = tid >> 2, q = tid & 3;
    const float* kprow = kp + ((size_t)s * Bc + b) * HDc;
    const float* qprow = qp + (size_t)b * HDc;
    float sum = 0.0f;
    for (int j = q * 128; j < q * 128 + 128; ++j)
        sum += vv[j] * tanhf(kprow[j] + qprow[j]);
    part[tid] = sum;
    __syncthreads();
    if (q == 0) {
        float e = part[tid] + part[tid + 1] + part[tid + 2] + part[tid + 3];
        sh_e[s] = (s < lens[b]) ? e : -10000000000.0f;
    }
    __syncthreads();
    if (tid == 0) {   // softmax over S=64, serial (tiny)
        float mx = -3.4e38f;
        for (int i = 0; i < Sc; ++i) mx = fmaxf(mx, sh_e[i]);
        float den = 0.0f;
        for (int i = 0; i < Sc; ++i) { float w = expf(sh_e[i] - mx); sw[i] = w; den += w; }
        float inv = 1.0f / den;
        for (int i = 0; i < Sc; ++i) sw[i] *= inv;
    }
    __syncthreads();
    for (int d = tid; d < HDc; d += 256) {
        float ctx = 0.0f;
        for (int i = 0; i < Sc; ++i)
            ctx += sw[i] * bf2f(enc_bf[((size_t)i * Bc + b) * (2 * Hc) + d]);
        a_bf[((size_t)t * Bc + b) * HDc + d] = f2bf(h_d[(size_t)b * HDc + d] + ctx);
    }
}

// masked NLL: per (b,t) row of V logits -> logsumexp, accumulate num/den
__global__ void loss_kernel(const float* __restrict__ logits, const int* __restrict__ oseq,
                            float* __restrict__ accum) {
    int row = blockIdx.x;   // = b*T + t, matches both logits row and oseq flat index
    const float* lr = logits + (size_t)row * Vc;
    __shared__ float sm[256];
    int tid = threadIdx.x;
    float mx = -3.4e38f;
    for (int i = tid; i < Vc; i += 256) mx = fmaxf(mx, lr[i]);
    sm[tid] = mx; __syncthreads();
    for (int off = 128; off > 0; off >>= 1) {
        if (tid < off) sm[tid] = fmaxf(sm[tid], sm[tid + off]);
        __syncthreads();
    }
    mx = sm[0]; __syncthreads();
    float s = 0.0f;
    for (int i = tid; i < Vc; i += 256) s += expf(lr[i] - mx);
    sm[tid] = s; __syncthreads();
    for (int off = 128; off > 0; off >>= 1) {
        if (tid < off) sm[tid] += sm[tid + off];
        __syncthreads();
    }
    if (tid == 0) {
        int tgt = oseq[row];
        if (tgt != 0) {
            float lse = mx + logf(sm[0]);
            atomicAdd(accum + 0, lse - lr[tgt]);
            atomicAdd(accum + 1, 1.0f);
        }
    }
}

__global__ void finalize_kernel(const float* __restrict__ accum, float* __restrict__ out) {
    out[0] = accum[0] / fmaxf(accum[1], 1.0f);
}

// ---------------- host orchestration ----------------
static inline void launch_gemm(const ushort_t* A, int lda, const ushort_t* W, int ldw,
                               float* C, int ldc, int M, int N, int K,
                               const float* bias, int remap, hipStream_t stream) {
    int nT = (N + 63) / 64;
    int tiles = (M / 32) * nT;
    int blocks = (tiles + 7) / 8;   // 8 waves (256 threads) per workgroup
    if (K == 128) {
        gemm_bf16_kernel<128><<<blocks, 256, 0, stream>>>(A, lda, W, ldw, C, ldc, M, N, bias, nT, remap);
    } else if (K == 256) {
        gemm_bf16_kernel<256><<<blocks, 256, 0, stream>>>(A, lda, W, ldw, C, ldc, M, N, bias, nT, remap);
    } else {
        gemm_bf16_kernel<512><<<blocks, 256, 0, stream>>>(A, lda, W, ldw, C, ldc, M, N, bias, nT, remap);
    }
}

static inline void launch_cvt(const float* src, ushort_t* dst, int n, hipStream_t stream) {
    f32_to_bf16_kernel<<<(n + 255) / 256, 256, 0, stream>>>(src, dst, n);
}

extern "C" void kernel_launch(void* const* d_in, const int* in_sizes, int n_in,
                              void* d_out, int out_size, void* d_ws, size_t ws_size,
                              hipStream_t stream) {
    (void)in_sizes; (void)n_in; (void)out_size; (void)ws_size;

    const int*   seq_in  = (const int*)  d_in[0];
    const int*   lens    = (const int*)  d_in[1];
    const int*   seq_out = (const int*)  d_in[2];
    const float* emb_src = (const float*)d_in[3];
    const float* Wih_f   = (const float*)d_in[4];
    const float* Whh_f   = (const float*)d_in[5];
    const float* b_f     = (const float*)d_in[6];
    const float* Wih_b   = (const float*)d_in[7];
    const float* Whh_b   = (const float*)d_in[8];
    const float* b_b     = (const float*)d_in[9];
    const float* emb_tgt = (const float*)d_in[10];
    const float* Wih_d   = (const float*)d_in[11];
    const float* Whh_d   = (const float*)d_in[12];
    const float* b_d     = (const float*)d_in[13];
    const float* W1      = (const float*)d_in[14];
    const float* W2      = (const float*)d_in[15];
    const float* vvec    = (const float*)d_in[16];
    const float* Wo      = (const float*)d_in[17];
    const float* bo      = (const float*)d_in[18];

    float* out = (float*)d_out;

    // ---- workspace carve (256B aligned) ----
    char* p = (char*)d_ws;
    auto alloc = [&](size_t bytes) -> char* {
        char* r = p;
        p += (bytes + 255) & ~(size_t)255;
        return r;
    };
    ushort_t* Wih_f_bf = (ushort_t*)alloc((size_t)4 * Hc  * Ec  * 2);
    ushort_t* Whh_f_bf = (ushort_t*)alloc((size_t)4 * Hc  * Hc  * 2);
    ushort_t* Wih_b_bf = (ushort_t*)alloc((size_t)4 * Hc  * Ec  * 2);
    ushort_t* Whh_b_bf = (ushort_t*)alloc((size_t)4 * Hc  * Hc  * 2);
    ushort_t* Wih_d_bf = (ushort_t*)alloc((size_t)4 * HDc * Ec  * 2);
    ushort_t* Whh_d_bf = (ushort_t*)alloc((size_t)4 * HDc * HDc * 2);
    ushort_t* W1_bf    = (ushort_t*)alloc((size_t)HDc * HDc * 2);
    ushort_t* W2_bf    = (ushort_t*)alloc((size_t)HDc * HDc * 2);
    ushort_t* Wo_bf    = (ushort_t*)alloc((size_t)Vc  * HDc * 2);
    ushort_t* x_src_bf = (ushort_t*)alloc((size_t)Sc * Bc * Ec * 2);
    ushort_t* dec_bf   = (ushort_t*)alloc((size_t)Tc * Bc * Ec * 2);
    ushort_t* enc_bf   = (ushort_t*)alloc((size_t)Sc * Bc * (2 * Hc) * 2);
    ushort_t* a_bf     = (ushort_t*)alloc((size_t)Tc * Bc * HDc * 2);
    ushort_t* h_f_bf   = (ushort_t*)alloc((size_t)Bc * Hc  * 2);
    ushort_t* h_b_bf   = (ushort_t*)alloc((size_t)Bc * Hc  * 2);
    ushort_t* h_d_bf   = (ushort_t*)alloc((size_t)Bc * HDc * 2);
    float* xg_f = (float*)alloc((size_t)Sc * Bc * (4 * Hc)  * 4);
    float* xg_b = (float*)alloc((size_t)Sc * Bc * (4 * Hc)  * 4);
    float* xg_d = (float*)alloc((size_t)Tc * Bc * (4 * HDc) * 4);
    float* hg_f = (float*)alloc((size_t)Bc * (4 * Hc)  * 4);
    float* hg_b = (float*)alloc((size_t)Bc * (4 * Hc)  * 4);
    float* hg_d = (float*)alloc((size_t)Bc * (4 * HDc) * 4);
    float* h_f  = (float*)alloc((size_t)Bc * Hc  * 4);
    float* c_f  = (float*)alloc((size_t)Bc * Hc  * 4);
    float* h_b  = (float*)alloc((size_t)Bc * Hc  * 4);
    float* c_b  = (float*)alloc((size_t)Bc * Hc  * 4);
    float* h_d  = (float*)alloc((size_t)Bc * HDc * 4);
    float* c_d  = (float*)alloc((size_t)Bc * HDc * 4);
    float* kp   = (float*)alloc((size_t)Sc * Bc * HDc * 4);
    float* qp   = (float*)alloc((size_t)Bc * HDc * 4);
    float* accum = (float*)alloc(2 * 4);

    // ---- one-time conversions (fp32 -> bf16) ----
    launch_cvt(Wih_f, Wih_f_bf, 4 * Hc * Ec, stream);
    launch_cvt(Whh_f, Whh_f_bf, 4 * Hc * Hc, stream);
    launch_cvt(Wih_b, Wih_b_bf, 4 * Hc * Ec, stream);
    launch_cvt(Whh_b, Whh_b_bf, 4 * Hc * Hc, stream);
    launch_cvt(Wih_d, Wih_d_bf, 4 * HDc * Ec, stream);
    launch_cvt(Whh_d, Whh_d_bf, 4 * HDc * HDc, stream);
    launch_cvt(W1, W1_bf, HDc * HDc, stream);
    launch_cvt(W2, W2_bf, HDc * HDc, stream);
    launch_cvt(Wo, Wo_bf, Vc * HDc, stream);

    gather_src_kernel<<<(Sc * Bc * Ec + 255) / 256, 256, 0, stream>>>(emb_src, seq_in, x_src_bf);
    gather_dec_kernel<<<(Tc * Bc * Ec + 255) / 256, 256, 0, stream>>>(emb_tgt, seq_out, dec_bf);

    // ---- hoisted input-side projections (big WMMA GEMMs) ----
    launch_gemm(x_src_bf, Ec, Wih_f_bf, Ec, xg_f, 4 * Hc,  Sc * Bc, 4 * Hc,  Ec, b_f, 0, stream);
    launch_gemm(x_src_bf, Ec, Wih_b_bf, Ec, xg_b, 4 * Hc,  Sc * Bc, 4 * Hc,  Ec, b_b, 0, stream);
    launch_gemm(dec_bf,   Ec, Wih_d_bf, Ec, xg_d, 4 * HDc, Tc * Bc, 4 * HDc, Ec, b_d, 0, stream);

    // ---- zero recurrent states ----
    hipMemsetAsync(h_f, 0, (size_t)Bc * Hc * 4, stream);
    hipMemsetAsync(c_f, 0, (size_t)Bc * Hc * 4, stream);
    hipMemsetAsync(h_b, 0, (size_t)Bc * Hc * 4, stream);
    hipMemsetAsync(c_b, 0, (size_t)Bc * Hc * 4, stream);
    hipMemsetAsync(h_d, 0, (size_t)Bc * HDc * 4, stream);
    hipMemsetAsync(c_d, 0, (size_t)Bc * HDc * 4, stream);
    hipMemsetAsync(h_f_bf, 0, (size_t)Bc * Hc * 2, stream);
    hipMemsetAsync(h_b_bf, 0, (size_t)Bc * Hc * 2, stream);
    hipMemsetAsync(h_d_bf, 0, (size_t)Bc * HDc * 2, stream);
    hipMemsetAsync(accum, 0, 2 * 4, stream);

    // ---- encoder scans (fwd + bwd interleaved on the stream) ----
    for (int s = 0; s < Sc; ++s) {
        launch_gemm(h_f_bf, Hc, Whh_f_bf, Hc, hg_f, 4 * Hc, Bc, 4 * Hc, Hc, nullptr, 0, stream);
        lstm_cell_enc_kernel<<<(Bc * Hc + 255) / 256, 256, 0, stream>>>(
            xg_f, hg_f, h_f, c_f, h_f_bf, enc_bf, lens, s, 0);
        int sb = Sc - 1 - s;
        launch_gemm(h_b_bf, Hc, Whh_b_bf, Hc, hg_b, 4 * Hc, Bc, 4 * Hc, Hc, nullptr, 0, stream);
        lstm_cell_enc_kernel<<<(Bc * Hc + 255) / 256, 256, 0, stream>>>(
            xg_b, hg_b, h_b, c_b, h_b_bf, enc_bf, lens, sb, 1);
    }

    // ---- key_proj = enc @ W1^T ----
    launch_gemm(enc_bf, 2 * Hc, W1_bf, 2 * Hc, kp, HDc, Sc * Bc, HDc, 2 * Hc, nullptr, 0, stream);

    // ---- decoder scan: Whh GEMM -> cell -> W2 GEMM -> attention (h+ctx stored bf16) ----
    for (int t = 0; t < Tc; ++t) {
        launch_gemm(h_d_bf, HDc, Whh_d_bf, HDc, hg_d, 4 * HDc, Bc, 4 * HDc, HDc, nullptr, 0, stream);
        lstm_cell_dec_kernel<<<(Bc * HDc + 255) / 256, 256, 0, stream>>>(
            xg_d, hg_d, h_d, c_d, h_d_bf, t);
        launch_gemm(h_d_bf, HDc, W2_bf, HDc, qp, HDc, Bc, HDc, HDc, nullptr, 0, stream);
        attention_kernel<<<Bc, 256, 0, stream>>>(kp, qp, enc_bf, vvec, h_d, lens, a_bf, t);
    }

    // ---- dominant GEMM: logits(B*T, V) = (h+ctx) @ Wo^T + bo, scattered to (b,t) order ----
    launch_gemm(a_bf, HDc, Wo_bf, HDc, out, Vc, Tc * Bc, Vc, HDc, bo, 1, stream);

    // ---- masked NLL loss ----
    loss_kernel<<<Bc * Tc, 256, 0, stream>>>(out, seq_out, accum);
    finalize_kernel<<<1, 1, 0, stream>>>(accum, out + (size_t)Bc * Tc * Vc);
}
